// CausalSelfAttention_87162066305655
// MI455X (gfx1250) — matrix operand order repaired and checked by reading, not compiled, not run on hardware
//
#include <hip/hip_runtime.h>

// ---------------------------------------------------------------------------
// Types for CDNA5 WMMA (wave32) + TDM descriptors
// ---------------------------------------------------------------------------
typedef __attribute__((ext_vector_type(16))) __bf16        v16bf;
typedef __attribute__((ext_vector_type(8)))  float         v8f;
typedef __attribute__((ext_vector_type(4)))  unsigned int  v4u;
typedef __attribute__((ext_vector_type(4)))  __bf16        v4bf;
typedef __attribute__((ext_vector_type(8)))  int           i32x8;
typedef __attribute__((ext_vector_type(4)))  int           i32x4;

#define LDS_AS __attribute__((address_space(3)))

struct V4U2 { v4u a, b; };

__device__ __forceinline__ v16bf join16(v4u lo, v4u hi) {
    V4U2 t{lo, hi};
    return __builtin_bit_cast(v16bf, t);
}

__device__ __forceinline__ v8f wmma_bf16(v16bf a, v16bf b, v8f c) {
    // D(f32 16x16) = A(bf16 16x32) * B(bf16 32x16) + C
    return __builtin_amdgcn_wmma_f32_16x16x32_bf16(
        /*neg_a=*/false, a, /*neg_b=*/false, b,
        /*c_mod=*/(short)0, c, /*reuse_a=*/false, /*reuse_b=*/false);
}

// Generic pointer into LDS -> 32-bit LDS byte offset (addr[31:0] per aperture map)
__device__ __forceinline__ unsigned lds_off(const void* p) {
    return (unsigned)(unsigned long long)p;
}

// Force the DS path: build an addrspace(3) pointer from the LDS byte offset,
// so 16B reads lower to ds_load_b128 (not flat_load_b128).
__device__ __forceinline__ v4u lds_read16(const __bf16* p) {
    const LDS_AS v4u* lp = (const LDS_AS v4u*)(unsigned long long)lds_off(p);
    return *lp;
}
__device__ __forceinline__ void lds_store_bf16(__bf16* p, __bf16 v) {
    LDS_AS __bf16* lp = (LDS_AS __bf16*)(unsigned long long)lds_off(p);
    *lp = v;
}

// ---------------------------------------------------------------------------
// TDM: DMA a [128 rows x 32 bf16] tile from global (row stride = strideElems)
// into LDS with 16B padding after every 64B row => LDS row stride 80B (LDSW=40).
// Descriptor layout per CDNA5 ISA 8.3/8.4 (2-D tile, groups 2/3 unused).
// This toolchain's builtin takes 6 args (g0, g1, g2, g3, extra8, cpol).
// ---------------------------------------------------------------------------
#define LDSW 40                 // bf16 elements per LDS row (32 data + 8 pad)
#define TILE_ELEMS (128 * LDSW) // elements per LDS tile buffer
#define TILE_BYTES (TILE_ELEMS * 2)   // 10240 B

__device__ __forceinline__
void tdm_load_tile(const __bf16* gsrc, unsigned lds_byte, int strideElems) {
    unsigned long long ga = (unsigned long long)(size_t)gsrc;
    v4u g0;
    g0[0] = 1u;                                   // count=1 (valid), user mode
    g0[1] = lds_byte;                             // lds_addr
    g0[2] = (unsigned)ga;                         // global_addr[31:0]
    g0[3] = (unsigned)((ga >> 32) & 0x1FFFFFFu)   // global_addr[56:32]
            | (2u << 30);                         // type=2 ("image")
    i32x8 g1;
    g1[0] = (1 << 16)      // data_size = 1 -> 2 bytes (bf16)
          | (1 << 20)      // pad_enable
          | (3 << 22)      // pad_interval: 16 DWORDs = 64B before padding
          | (3 << 25);     // pad_amount: 4 DWORDs = 16B of padding
    g1[1] = 0;                                    // atomic_addr=0 | tensor_dim0 lo16 (0)
    g1[2] = 16 | (0 << 16);                       // tensor_dim0 = 1<<20 (hi16=16) | tensor_dim1 lo16
    g1[3] = 16 | (32 << 16);                      // tensor_dim1 = 1<<20 (hi16=16) | tile_dim0 = 32
    g1[4] = 128;                                  // tile_dim1 = 128 rows (tile_dim2 = 0)
    g1[5] = strideElems;                          // tensor_dim0_stride[31:0] (elements)
    g1[6] = 0;                                    // stride hi | dim1_stride lo
    g1[7] = 0;
    i32x4 gz4 = {0, 0, 0, 0};                     // groups 2/3 unused (2-D tile)
    i32x8 gz8 = {0, 0, 0, 0, 0, 0, 0, 0};
    __builtin_amdgcn_tensor_load_to_lds(g0, g1, gz4, gz4, gz8, 0);
}

// ---------------------------------------------------------------------------
// Elementwise f32 -> bf16
// ---------------------------------------------------------------------------
__global__ __launch_bounds__(256)
void cvt_f32_bf16(const float* __restrict__ in, __bf16* __restrict__ out, int n) {
    int i = (blockIdx.x * 256 + threadIdx.x) * 4;
    if (i + 3 < n) {
        float4 f = *(const float4*)(in + i);
        v4bf o;
        o[0] = (__bf16)f.x; o[1] = (__bf16)f.y; o[2] = (__bf16)f.z; o[3] = (__bf16)f.w;
        *(v4bf*)(out + i) = o;
    }
}

// w: [K][N] f32 row-major  ->  wt: [N][K] bf16 row-major
__global__ __launch_bounds__(256)
void transpose_cvt(const float* __restrict__ w, __bf16* __restrict__ wt, int K, int N) {
    int n = blockIdx.x * 256 + threadIdx.x;
    int k = blockIdx.y;
    if (n < N) wt[(size_t)n * K + k] = (__bf16)w[(size_t)k * N + n];
}

// ---------------------------------------------------------------------------
// Shared GEMM mainloop: C[128x128] += A[128xK] * Bt[128xK]^T, bf16 in, f32 acc.
// Double-buffered TDM staging (wave 0 issues tensor_load_to_lds for stage i+1,
// waits TENSORcnt<=2 so stage i is resident), K-step 32, 8 waves in 2(M)x4(N),
// each wave 4x2 16x16 WMMA tiles. AsD/BsD are double-length shared buffers.
// ---------------------------------------------------------------------------
__device__ __forceinline__
void gemm_mainloop(const __bf16* __restrict__ A, const __bf16* __restrict__ Bt,
                   int K, int m0, int n0,
                   __bf16* AsD, __bf16* BsD, v8f acc[4][2], int tid) {
    const int wave = tid >> 5, lane = tid & 31;
    const int lh = lane & 15, hig = lane >> 4;
    const int wm = wave >> 2, wn = wave & 3;
    const int nIter = K >> 5;

    const __bf16* Agbl = A + (size_t)m0 * K;
    const __bf16* Bgbl = Bt + (size_t)n0 * K;
    const unsigned asB = lds_off(AsD), bsB = lds_off(BsD);

    if (wave == 0) {
        tdm_load_tile(Agbl, asB, K);
        tdm_load_tile(Bgbl, bsB, K);
    }

    for (int it = 0; it < nIter; ++it) {
        if (wave == 0) {
            if (it + 1 < nIter) {
                unsigned nb = ((unsigned)(it + 1) & 1u) * TILE_BYTES;
                tdm_load_tile(Agbl + (it + 1) * 32, asB + nb, K);
                tdm_load_tile(Bgbl + (it + 1) * 32, bsB + nb, K);
                __builtin_amdgcn_s_wait_tensorcnt(2);   // stage `it` resident
            } else {
                __builtin_amdgcn_s_wait_tensorcnt(0);
            }
        }
        __syncthreads();

        const int cb = (it & 1) * TILE_ELEMS;           // current buffer (elements)
        const __bf16* As = AsD + cb;
        const __bf16* Bs = BsD + cb;
        v16bf af[4], bfr[2];
#pragma unroll
        for (int i = 0; i < 4; i++) {
            // A-frag (16x32): lanes<16 hold k0..15, lanes>=16 hold k16..31 (contiguous)
            const __bf16* p = As + (wm * 64 + i * 16 + lh) * LDSW + (hig ? 16 : 0);
            af[i] = join16(lds_read16(p), lds_read16(p + 8));
        }
#pragma unroll
        for (int j = 0; j < 2; j++) {
            // B-frag (32x16): lanes<16 hold k0..15, lanes>=16 hold k16..31 (contiguous)
            const __bf16* p = Bs + (wn * 32 + j * 16 + lh) * LDSW + (hig ? 16 : 0);
            bfr[j] = join16(lds_read16(p), lds_read16(p + 8));
        }
#pragma unroll
        for (int i = 0; i < 4; i++)
#pragma unroll
            for (int j = 0; j < 2; j++)
                acc[i][j] = wmma_bf16(af[i], bfr[j], acc[i][j]);

        asm volatile("s_wait_dscnt 0" ::: "memory");    // reads done before reuse
        __syncthreads();
    }
}

// ---------------------------------------------------------------------------
// GEMM 1: qkv = x @ w_attn + b_attn, scattered to attention-friendly layouts:
//   q,k -> [B,H,T,D] bf16 ; v -> [B,H,D,T] bf16 (transposed for P@V B-frags)
// ---------------------------------------------------------------------------
__global__ __launch_bounds__(256)
void gemm_qkv(const __bf16* __restrict__ xb, const __bf16* __restrict__ waT,
              const float* __restrict__ bias,
              __bf16* __restrict__ qb, __bf16* __restrict__ kb, __bf16* __restrict__ vt) {
    __shared__ __attribute__((aligned(16))) __bf16 AsD[2 * TILE_ELEMS];
    __shared__ __attribute__((aligned(16))) __bf16 BsD[2 * TILE_ELEMS];
    const int m0 = blockIdx.y * 128, n0 = blockIdx.x * 128;
    const int tid = threadIdx.x;
    v8f acc[4][2] = {};
    gemm_mainloop(xb, waT, 1024, m0, n0, AsD, BsD, acc, tid);

    const int wave = tid >> 5, lane = tid & 31;
    const int lh = lane & 15, hig = lane >> 4;
    const int wm = wave >> 2, wn = wave & 3;
#pragma unroll
    for (int i = 0; i < 4; i++)
#pragma unroll
        for (int j = 0; j < 2; j++) {
            int ncol = n0 + wn * 32 + j * 16 + lh;          // [0,3072)
            int sel = ncol >> 10, c = ncol & 1023;
            int h = c >> 6, d = c & 63;
            float bv = bias[ncol];
#pragma unroll
            for (int e = 0; e < 8; e++) {
                int mrow = m0 + wm * 64 + i * 16 + e + hig * 8;   // [0,4096)
                int b = mrow >> 11, t = mrow & 2047;
                int bh = b * 16 + h;
                float vvl = acc[i][j][e] + bv;
                if (sel == 0)      qb[((size_t)bh * 2048 + t) * 64 + d] = (__bf16)vvl;
                else if (sel == 1) kb[((size_t)bh * 2048 + t) * 64 + d] = (__bf16)vvl;
                else               vt[((size_t)bh * 64 + d) * 2048 + t] = (__bf16)vvl;
            }
        }
}

// ---------------------------------------------------------------------------
// Flash attention: per wave = one 16-row q-tile, online softmax (f32),
// S = Q K^T and O += P V via bf16 WMMA, P reshaped through per-wave LDS.
// Writes y as bf16 [B,T,C] for the proj GEMM.
// ---------------------------------------------------------------------------
__global__ __launch_bounds__(256)
void flash_attn(const __bf16* __restrict__ qb, const __bf16* __restrict__ kb,
                const __bf16* __restrict__ vt, __bf16* __restrict__ yb) {
    __shared__ __attribute__((aligned(16))) __bf16 Pl[8 * 16 * 32];
    const int bh = blockIdx.x;                 // 0..31
    const int b = bh >> 4, h = bh & 15;
    const int wave = threadIdx.x >> 5, lane = threadIdx.x & 31;
    const int lh = lane & 15, hig = lane >> 4;
    const int qlo = blockIdx.y * 128 + wave * 16;

    const __bf16* Q = qb + (size_t)bh * 2048 * 64;
    const __bf16* K = kb + (size_t)bh * 2048 * 64;
    const __bf16* V = vt + (size_t)bh * 64 * 2048;
    __bf16* Pw = Pl + wave * (16 * 32);

    // Q fragments for d-chunks {0..31},{32..63}
    v16bf qf[2];
    const int qr = qlo + lh;
#pragma unroll
    for (int c = 0; c < 2; c++) {
        const __bf16* p = Q + (size_t)qr * 64 + c * 32 + (hig ? 16 : 0);
        qf[c] = join16(*(const v4u*)p, *(const v4u*)(p + 8));
    }

    v8f o[4] = {};
    float m[8], l[8];
#pragma unroll
    for (int i = 0; i < 8; i++) { m[i] = -__builtin_inff(); l[i] = 0.0f; }
    const float sc = 0.125f;    // 1/sqrt(64)

    for (int kk = 0; kk < qlo + 16; kk += 32) {
        // S tiles for keys [kk, kk+16) and [kk+16, kk+32)
        v8f s0 = {}, s1 = {};
#pragma unroll
        for (int c = 0; c < 2; c++) {
            const __bf16* p0 = K + (size_t)(kk + lh) * 64 + c * 32 + (hig ? 16 : 0);
            v16bf kf0 = join16(*(const v4u*)p0, *(const v4u*)(p0 + 8));
            s0 = wmma_bf16(qf[c], kf0, s0);
            const __bf16* p1 = K + (size_t)(kk + 16 + lh) * 64 + c * 32 + (hig ? 16 : 0);
            v16bf kf1 = join16(*(const v4u*)p1, *(const v4u*)(p1 + 8));
            s1 = wmma_bf16(qf[c], kf1, s1);
        }
        // Causal mask + online softmax (row r lives in a 16-lane half)
        const int key0 = kk + lh, key1 = key0 + 16;
#pragma unroll
        for (int i = 0; i < 8; i++) {
            int q = qlo + i + hig * 8;
            float a0 = (key0 <= q) ? s0[i] * sc : -__builtin_inff();
            float a1 = (key1 <= q) ? s1[i] * sc : -__builtin_inff();
            float rm = fmaxf(a0, a1);
#pragma unroll
            for (int off = 8; off; off >>= 1) rm = fmaxf(rm, __shfl_xor(rm, off, 32));
            float mn = fmaxf(m[i], rm);
            float alpha = __expf(m[i] - mn);
            float p0 = __expf(a0 - mn);
            float p1 = __expf(a1 - mn);
            float rs = p0 + p1;
#pragma unroll
            for (int off = 8; off; off >>= 1) rs += __shfl_xor(rs, off, 32);
            l[i] = l[i] * alpha + rs;
            m[i] = mn;
#pragma unroll
            for (int j = 0; j < 4; j++) o[j][i] *= alpha;
            int r = i + hig * 8;
            lds_store_bf16(Pw + r * 32 + lh,      (__bf16)p0);  // C-layout -> row-major P
            lds_store_bf16(Pw + r * 32 + 16 + lh, (__bf16)p1);
        }
        asm volatile("s_wait_dscnt 0" ::: "memory");   // P stores visible to frag loads

        // P fragment (A 16x32) from LDS, then O += P @ V  (4 d-tiles)
        const __bf16* pp = Pw + lh * 32 + (hig ? 16 : 0);
        v16bf pf = join16(lds_read16(pp), lds_read16(pp + 8));
#pragma unroll
        for (int j = 0; j < 4; j++) {
            const __bf16* vp = V + (size_t)(j * 16 + lh) * 2048 + kk + (hig ? 16 : 0);
            v16bf vf = join16(*(const v4u*)vp, *(const v4u*)(vp + 8));
            o[j] = wmma_bf16(pf, vf, o[j]);
        }
    }

    // Normalize and scatter y[b, t, h*64 + d] (bf16)
#pragma unroll
    for (int i = 0; i < 8; i++) {
        float inv = 1.0f / l[i];
        int q = qlo + i + hig * 8;
        __bf16* yrow = yb + ((size_t)(b * 2048 + q) * 1024) + h * 64;
#pragma unroll
        for (int j = 0; j < 4; j++)
            yrow[j * 16 + lh] = (__bf16)(o[j][i] * inv);
    }
}

// ---------------------------------------------------------------------------
// GEMM 2: out = y @ w_proj + b_proj  (f32 output)
// ---------------------------------------------------------------------------
__global__ __launch_bounds__(256)
void gemm_proj(const __bf16* __restrict__ yb, const __bf16* __restrict__ wpT,
               const float* __restrict__ bias, float* __restrict__ out) {
    __shared__ __attribute__((aligned(16))) __bf16 AsD[2 * TILE_ELEMS];
    __shared__ __attribute__((aligned(16))) __bf16 BsD[2 * TILE_ELEMS];
    const int m0 = blockIdx.y * 128, n0 = blockIdx.x * 128;
    const int tid = threadIdx.x;
    v8f acc[4][2] = {};
    gemm_mainloop(yb, wpT, 1024, m0, n0, AsD, BsD, acc, tid);

    const int wave = tid >> 5, lane = tid & 31;
    const int lh = lane & 15, hig = lane >> 4;
    const int wm = wave >> 2, wn = wave & 3;
#pragma unroll
    for (int i = 0; i < 4; i++)
#pragma unroll
        for (int j = 0; j < 2; j++) {
            int ncol = n0 + wn * 32 + j * 16 + lh;
            float bv = bias[ncol];
#pragma unroll
            for (int e = 0; e < 8; e++) {
                int mrow = m0 + wm * 64 + i * 16 + e + hig * 8;
                out[(size_t)mrow * 1024 + ncol] = acc[i][j][e] + bv;
            }
        }
}

// ---------------------------------------------------------------------------
// Launch: B=2, T=2048, C=1024, H=16, D=64
// Workspace layout (40 MB total):
//   [ 0MB) waT  bf16 [3072][1024]   (6MB)
//   [ 6MB) wpT  bf16 [1024][1024]   (2MB)
//   [ 8MB) xb   bf16 [4096][1024]   (8MB)  -- reused as yb after qkv GEMM
//   [16MB) qb   bf16 [B,H,T,D]      (8MB)
//   [24MB) kb   bf16 [B,H,T,D]      (8MB)
//   [32MB) vt   bf16 [B,H,D,T]      (8MB)
// ---------------------------------------------------------------------------
extern "C" void kernel_launch(void* const* d_in, const int* in_sizes, int n_in,
                              void* d_out, int out_size, void* d_ws, size_t ws_size,
                              hipStream_t stream) {
    const float* x      = (const float*)d_in[0];
    const float* w_attn = (const float*)d_in[1];
    const float* b_attn = (const float*)d_in[2];
    const float* w_proj = (const float*)d_in[3];
    const float* b_proj = (const float*)d_in[4];

    char* ws = (char*)d_ws;
    const size_t MB = 1024u * 1024u;
    __bf16* waT = (__bf16*)(ws + 0 * MB);
    __bf16* wpT = (__bf16*)(ws + 6 * MB);
    __bf16* xb  = (__bf16*)(ws + 8 * MB);
    __bf16* qb  = (__bf16*)(ws + 16 * MB);
    __bf16* kb  = (__bf16*)(ws + 24 * MB);
    __bf16* vt  = (__bf16*)(ws + 32 * MB);
    __bf16* yb  = xb;   // x's bf16 copy is dead after the qkv GEMM

    // 1) precision conversions / weight transposes
    cvt_f32_bf16<<<4096, 256, 0, stream>>>(x, xb, 4096 * 1024);
    transpose_cvt<<<dim3(12, 1024), 256, 0, stream>>>(w_attn, waT, 1024, 3072);
    transpose_cvt<<<dim3(4, 1024), 256, 0, stream>>>(w_proj, wpT, 1024, 1024);

    // 2) qkv = x @ w_attn + b_attn  -> q,k [B,H,T,D], v [B,H,D,T]
    gemm_qkv<<<dim3(24, 32), 256, 0, stream>>>(xb, waT, b_attn, qb, kb, vt);

    // 3) causal flash attention -> y bf16 [B,T,C]
    flash_attn<<<dim3(32, 16), 256, 0, stream>>>(qb, kb, vt, yb);

    // 4) out = y @ w_proj + b_proj (f32)
    gemm_proj<<<dim3(8, 32), 256, 0, stream>>>(yb, wpT, b_proj, (float*)d_out);
}
